// PseudoLabelGenerator_47236050321907
// MI455X (gfx1250) — matrix operand order repaired and checked by reading, not compiled, not run on hardware
//
#include <hip/hip_runtime.h>

typedef __attribute__((ext_vector_type(16))) _Float16 v16h;
typedef __attribute__((ext_vector_type(8)))  float    v8f;

namespace {
constexpr int N_     = 100000;
constexpr int E_     = 3200000;
constexpr int C_     = 10;
constexpr int HID_   = 256;
constexpr int LH_    = 128;     // LSTM hidden
constexpr int G4_    = 4 * LH_; // 512 gate width
constexpr int T_     = 12;
constexpr int HEADS_ = 4;

// float32 workspace layout (element offsets)
constexpr size_t OFF_PROBS = 0;                         // N*10
constexpr size_t OFF_X     = OFF_PROBS + (size_t)N_*10; // N*40
constexpr size_t OFF_AS    = OFF_X     + (size_t)N_*40; // N*4
constexpr size_t OFF_AD    = OFF_AS    + (size_t)N_*4;  // N*4
constexpr size_t OFF_M     = OFF_AD    + (size_t)N_*4;  // N*4
constexpr size_t OFF_S     = OFF_M     + (size_t)N_*4;  // N*4
constexpr size_t OFF_GOUT  = OFF_S     + (size_t)N_*4;  // N*40
constexpr size_t OFF_DEG   = OFF_GOUT  + (size_t)N_*40; // N
constexpr size_t OFF_NSUM  = OFF_DEG   + (size_t)N_;    // N*10
constexpr size_t OFF_F32_END = OFF_NSUM + (size_t)N_*10;

// fp16 weight region (element offsets within fp16 area)
constexpr size_t HOFF_CW1  = 0;                       // 128*256
constexpr size_t HOFF_WHH0 = HOFF_CW1  + 128*256;     // 512*128
constexpr size_t HOFF_WIH1 = HOFF_WHH0 + 512*128;     // 512*128
constexpr size_t HOFF_WHH1 = HOFF_WIH1 + 512*128;     // 512*128
} // namespace

__device__ __forceinline__ float sigm(float x) { return 1.0f / (1.0f + __expf(-x)); }

__device__ __forceinline__ v8f zero8() {
  v8f z = {0.f, 0.f, 0.f, 0.f, 0.f, 0.f, 0.f, 0.f};
  return z;
}

__device__ __forceinline__ v8f wmma_f16(v16h a, v16h b, v8f c) {
  // (neg_a, A, neg_b, B, c_mod, C, reuse_a, reuse_b)
  return __builtin_amdgcn_wmma_f32_16x16x32_f16(false, a, false, b, (short)0, c, false, false);
}

// A fragment: 16x32 f16, row-major source with leading dim lda (LDS or global).
// Lane l holds row m = l&15; group g = l>>4; VGPR v half h -> k = 16g+2v+h.
__device__ __forceinline__ v16h load_a_rm(const _Float16* src, int lda, int kbase, int lane) {
  int m = lane & 15;
  int g = (lane >> 4) & 1;
  const _Float16* row = src + (size_t)m * lda + kbase;
  v16h a;
#pragma unroll
  for (int v = 0; v < 8; ++v) {
    int kk = 16 * g + 2 * v;
    a[2 * v]     = row[kk];
    a[2 * v + 1] = row[kk + 1];
  }
  return a;
}

// B fragment: 32x16 f16 where B[k][n] = W[n][k], W row-major [nDim][kDim], ld = kDim.
// Lane l holds column n = nbase + (l&15); group g = l>>4; VGPR v half h -> k = 16g+2v+h.
__device__ __forceinline__ v16h load_b_wt(const _Float16* W, int ldw, int nbase, int kbase, int lane) {
  int n = nbase + (lane & 15);
  int g = (lane >> 4) & 1;
  const _Float16* col = W + (size_t)n * ldw + kbase + 16 * g;
  v16h b;
#pragma unroll
  for (int v = 0; v < 8; ++v) {
    b[2 * v]     = col[2 * v];
    b[2 * v + 1] = col[2 * v + 1];
  }
  return b;
}

__device__ __forceinline__ void atomicMaxF32(float* addr, float v) {
  __hip_atomic_fetch_max(addr, v, __ATOMIC_RELAXED, __HIP_MEMORY_SCOPE_AGENT);
}

// ---------------------------------------------------------------------------
// f32 -> f16 weight conversion
// ---------------------------------------------------------------------------
__global__ void k_convert_f16(const float* __restrict__ src, _Float16* __restrict__ dst, int n) {
  int i = blockIdx.x * blockDim.x + threadIdx.x;
  if (i < n) dst[i] = (_Float16)src[i];
}

// ---------------------------------------------------------------------------
// Per-node: softmax, argmax, entropy, GAT node prep, accumulator init
// ---------------------------------------------------------------------------
__global__ void k_node(const float* __restrict__ pred,
                       const float* __restrict__ gW,
                       const float* __restrict__ gas,
                       const float* __restrict__ gad,
                       float* __restrict__ probs, float* __restrict__ x,
                       float* __restrict__ a_s, float* __restrict__ a_d,
                       float* __restrict__ m_, float* __restrict__ s_,
                       float* __restrict__ gout, float* __restrict__ deg,
                       float* __restrict__ nsum,
                       float* __restrict__ out_lab, float* __restrict__ out_maxp,
                       float* __restrict__ out_unc) {
  int n = blockIdx.x * blockDim.x + threadIdx.x;
  if (n >= N_) return;
  float p[C_];
  float mx = -3.0e38f;
#pragma unroll
  for (int c = 0; c < C_; ++c) { p[c] = pred[(size_t)n * C_ + c]; mx = fmaxf(mx, p[c]); }
  float sum = 0.f;
#pragma unroll
  for (int c = 0; c < C_; ++c) { p[c] = __expf(p[c] - mx); sum += p[c]; }
  float inv = 1.0f / sum;
  int arg = 0;
  float best = -1.f, ent = 0.f;
#pragma unroll
  for (int c = 0; c < C_; ++c) {
    p[c] *= inv;
    probs[(size_t)n * C_ + c] = p[c];
    if (p[c] > best) { best = p[c]; arg = c; }
    ent -= p[c] * __logf(p[c] + 1e-8f);
  }
  out_lab[n]  = (float)arg;
  out_maxp[n] = best;
  out_unc[n]  = ent * (1.0f / 2.302585093f); // / ln(10)

  // x = probs @ gW  -> [HEADS*C]
  float xv[HEADS_ * C_];
#pragma unroll
  for (int j = 0; j < HEADS_ * C_; ++j) {
    float a = 0.f;
#pragma unroll
    for (int k = 0; k < C_; ++k) a += p[k] * gW[k * (HEADS_ * C_) + j];
    xv[j] = a;
    x[(size_t)n * 40 + j] = a;
  }
#pragma unroll
  for (int h = 0; h < HEADS_; ++h) {
    float as = 0.f, ad = 0.f;
#pragma unroll
    for (int c = 0; c < C_; ++c) {
      as += xv[h * C_ + c] * gas[h * C_ + c];
      ad += xv[h * C_ + c] * gad[h * C_ + c];
    }
    a_s[(size_t)n * 4 + h] = as;
    a_d[(size_t)n * 4 + h] = ad;
    m_[(size_t)n * 4 + h] = -3.0e38f;
    s_[(size_t)n * 4 + h] = 0.f;
  }
#pragma unroll
  for (int j = 0; j < 40; ++j) gout[(size_t)n * 40 + j] = 0.f;
  deg[n] = 0.f;
#pragma unroll
  for (int c = 0; c < C_; ++c) nsum[(size_t)n * C_ + c] = 0.f;
}

// ---------------------------------------------------------------------------
// Confidence net: sigmoid( relu(emb @ cW1^T + cb1) @ cW2^T + cb2 )
// 16 nodes / block, 8 waves; wave w owns output columns [16w,16w+16)
// ---------------------------------------------------------------------------
__global__ __launch_bounds__(256) void k_confnet(const float* __restrict__ emb,
                                                 const float* __restrict__ cb1,
                                                 const float* __restrict__ cW2,
                                                 const float* __restrict__ cb2,
                                                 const _Float16* __restrict__ cW1h,
                                                 float* __restrict__ out_emb) {
  __shared__ _Float16 embh[16 * HID_]; // 8 KB
  __shared__ float    h1[16 * 128];    // 8 KB
  const int tid = threadIdx.x;
  const int node0 = blockIdx.x * 16;
  for (int p = tid; p < 16 * HID_; p += 256) {
    int m = p >> 8, k = p & 255;
    embh[p] = (_Float16)emb[(size_t)(node0 + m) * HID_ + k];
  }
  __syncthreads();
  const int lane = tid & 31, wave = tid >> 5;
  v8f acc = zero8();
#pragma unroll
  for (int kt = 0; kt < HID_; kt += 32) {
    v16h a = load_a_rm(embh, HID_, kt, lane);
    v16h b = load_b_wt(cW1h, HID_, wave * 16, kt, lane);
    acc = wmma_f16(a, b, acc);
  }
  const int mg = (lane >> 4) * 8;
  const int n  = wave * 16 + (lane & 15);
  const float bias = cb1[n];
#pragma unroll
  for (int v = 0; v < 8; ++v) {
    float xx = acc[v] + bias;
    h1[(mg + v) * 128 + n] = xx > 0.f ? xx : 0.f;
  }
  __syncthreads();
  if (tid < 16) {
    float s = cb2[0];
    for (int k = 0; k < 128; ++k) s += h1[tid * 128 + k] * cW2[k];
    out_emb[node0 + tid] = sigm(s);
  }
}

// ---------------------------------------------------------------------------
// 2-layer LSTM over T=12 steps (input constant per node), WMMA recurrent GEMMs.
// Block = 256 threads (8 waves) = 16 nodes. Wave w owns 4 N-tiles of the
// 512-wide gate matrix. Cell state / variance stats live in registers.
// ---------------------------------------------------------------------------
__global__ __launch_bounds__(256) void k_lstm(const float* __restrict__ pred,
                                              const float* __restrict__ w_ih0,
                                              const float* __restrict__ b_ih0,
                                              const float* __restrict__ b_hh0,
                                              const float* __restrict__ b_ih1,
                                              const float* __restrict__ b_hh1,
                                              const _Float16* __restrict__ whh0h,
                                              const _Float16* __restrict__ wih1h,
                                              const _Float16* __restrict__ whh1h,
                                              float* __restrict__ out_tc) {
  __shared__ _Float16 gx0h[16 * G4_];  // 16 KB: constant x-part of layer-0 gates
  __shared__ float    gates[16 * G4_]; // 32 KB
  __shared__ _Float16 h0h[16 * LH_];   // 4 KB
  __shared__ _Float16 h1h[16 * LH_];   // 4 KB
  __shared__ float    varsum[16];

  const int tid = threadIdx.x;
  const int node0 = blockIdx.x * 16;
  const int lane = tid & 31, wave = tid >> 5;

  // precompute gx0 = pred @ w_ih0^T + b_ih0 + b_hh0 (C=10 input, scalar path)
  for (int p = tid; p < 16 * G4_; p += 256) {
    int m = p >> 9, n = p & (G4_ - 1);
    float acc = b_ih0[n] + b_hh0[n];
    const float* pr = pred + (size_t)(node0 + m) * C_;
#pragma unroll
    for (int k = 0; k < C_; ++k) acc += pr[k] * w_ih0[n * C_ + k];
    gx0h[p] = (_Float16)acc;
  }
  for (int p = tid; p < 16 * LH_; p += 256) { h0h[p] = (_Float16)0.f; h1h[p] = (_Float16)0.f; }

  // fixed per-thread (m, j) mapping for element-wise phases
  const int em = tid >> 4;
  const int jb = (tid & 15) * 8;
  float c0r[8], c1r[8], sh[8], sq[8];
#pragma unroll
  for (int q = 0; q < 8; ++q) { c0r[q] = 0.f; c1r[q] = 0.f; sh[q] = 0.f; sq[q] = 0.f; }
  __syncthreads();

  for (int t = 0; t < T_; ++t) {
    // ---- layer 0: gates = gx0 + h0 @ whh0^T -------------------------------
    v8f acc[4];
#pragma unroll
    for (int i = 0; i < 4; ++i) acc[i] = zero8();
#pragma unroll
    for (int kt = 0; kt < LH_; kt += 32) {
      v16h a = load_a_rm(h0h, LH_, kt, lane);
#pragma unroll
      for (int i = 0; i < 4; ++i) {
        v16h b = load_b_wt(whh0h, LH_, (wave * 4 + i) * 16, kt, lane);
        acc[i] = wmma_f16(a, b, acc[i]);
      }
    }
    __syncthreads(); // gates free (previous layer-1 elementwise done reading)
    {
      const int mg = (lane >> 4) * 8;
      const int nl = lane & 15;
#pragma unroll
      for (int i = 0; i < 4; ++i) {
        int n = (wave * 4 + i) * 16 + nl;
#pragma unroll
        for (int v = 0; v < 8; ++v)
          gates[(mg + v) * G4_ + n] = acc[i][v] + (float)gx0h[(mg + v) * G4_ + n];
      }
    }
    __syncthreads();
    // ---- layer 0 element-wise --------------------------------------------
    {
      const float* gm = gates + em * G4_;
#pragma unroll
      for (int q = 0; q < 8; ++q) {
        int j = jb + q;
        float gi = gm[j], gf = gm[LH_ + j], gg = gm[2 * LH_ + j], go = gm[3 * LH_ + j];
        c0r[q] = sigm(gf) * c0r[q] + sigm(gi) * tanhf(gg);
        float hv = sigm(go) * tanhf(c0r[q]);
        h0h[em * LH_ + j] = (_Float16)hv;
      }
    }
    __syncthreads();
    // ---- layer 1: gates = h0 @ wih1^T + h1 @ whh1^T + b ------------------
#pragma unroll
    for (int i = 0; i < 4; ++i) acc[i] = zero8();
#pragma unroll
    for (int kt = 0; kt < LH_; kt += 32) {
      v16h a = load_a_rm(h0h, LH_, kt, lane);
#pragma unroll
      for (int i = 0; i < 4; ++i) {
        v16h b = load_b_wt(wih1h, LH_, (wave * 4 + i) * 16, kt, lane);
        acc[i] = wmma_f16(a, b, acc[i]);
      }
    }
#pragma unroll
    for (int kt = 0; kt < LH_; kt += 32) {
      v16h a = load_a_rm(h1h, LH_, kt, lane);
#pragma unroll
      for (int i = 0; i < 4; ++i) {
        v16h b = load_b_wt(whh1h, LH_, (wave * 4 + i) * 16, kt, lane);
        acc[i] = wmma_f16(a, b, acc[i]);
      }
    }
    {
      const int mg = (lane >> 4) * 8;
      const int nl = lane & 15;
#pragma unroll
      for (int i = 0; i < 4; ++i) {
        int n = (wave * 4 + i) * 16 + nl;
        float bias = b_ih1[n] + b_hh1[n];
#pragma unroll
        for (int v = 0; v < 8; ++v)
          gates[(mg + v) * G4_ + n] = acc[i][v] + bias;
      }
    }
    __syncthreads();
    // ---- layer 1 element-wise + variance stats ---------------------------
    {
      const float* gm = gates + em * G4_;
#pragma unroll
      for (int q = 0; q < 8; ++q) {
        int j = jb + q;
        float gi = gm[j], gf = gm[LH_ + j], gg = gm[2 * LH_ + j], go = gm[3 * LH_ + j];
        c1r[q] = sigm(gf) * c1r[q] + sigm(gi) * tanhf(gg);
        float hv = sigm(go) * tanhf(c1r[q]);
        h1h[em * LH_ + j] = (_Float16)hv;
        sh[q] += hv;
        sq[q] += hv * hv;
      }
    }
    __syncthreads();
  }

  // temporal variance (unbiased), mean over hidden dims
  float vs = 0.f;
#pragma unroll
  for (int q = 0; q < 8; ++q) {
    float mean = sh[q] * (1.0f / T_);
    float var = (sq[q] - (float)T_ * mean * mean) * (1.0f / (T_ - 1));
    vs += var;
  }
  if (tid < 16) varsum[tid] = 0.f;
  __syncthreads();
  atomicAdd(&varsum[em], vs);
  __syncthreads();
  if (tid < 16) {
    float tv = varsum[tid] * (1.0f / LH_);
    out_tc[node0 + tid] = 1.0f / (1.0f + tv);
  }
}

// ---------------------------------------------------------------------------
// GAT edge pass 1: per-dst per-head running max of leaky-relu attention logit;
// also degree / neighbor-label histograms for the agreement check.
// ---------------------------------------------------------------------------
__global__ void k_edge_max(const int* __restrict__ eidx,
                           const float* __restrict__ a_s, const float* __restrict__ a_d,
                           const float* __restrict__ out_lab,
                           float* __restrict__ m_, float* __restrict__ deg,
                           float* __restrict__ nsum) {
  int e = blockIdx.x * blockDim.x + threadIdx.x;
  const int total = E_ + N_;
  if (e >= total) return;
  int src, dst;
  if (e < E_) { src = eidx[e]; dst = eidx[E_ + e]; }
  else { src = e - E_; dst = e - E_; }
#pragma unroll
  for (int h = 0; h < HEADS_; ++h) {
    float v = a_s[(size_t)src * 4 + h] + a_d[(size_t)dst * 4 + h];
    v = (v > 0.f) ? v : 0.2f * v;
    atomicMaxF32(&m_[(size_t)dst * 4 + h], v);
  }
  atomicAdd(&deg[src], 1.0f);
  int lab = (int)out_lab[src];
  atomicAdd(&nsum[(size_t)dst * C_ + lab], 1.0f);
}

// GAT edge pass 2: softmax denominators.
__global__ void k_edge_sum(const int* __restrict__ eidx,
                           const float* __restrict__ a_s, const float* __restrict__ a_d,
                           const float* __restrict__ m_, float* __restrict__ s_) {
  int e = blockIdx.x * blockDim.x + threadIdx.x;
  const int total = E_ + N_;
  if (e >= total) return;
  int src, dst;
  if (e < E_) { src = eidx[e]; dst = eidx[E_ + e]; }
  else { src = e - E_; dst = e - E_; }
#pragma unroll
  for (int h = 0; h < HEADS_; ++h) {
    float v = a_s[(size_t)src * 4 + h] + a_d[(size_t)dst * 4 + h];
    v = (v > 0.f) ? v : 0.2f * v;
    atomicAdd(&s_[(size_t)dst * 4 + h], __expf(v - m_[(size_t)dst * 4 + h]));
  }
}

// GAT edge pass 3: weighted aggregation of x[src] into gout[dst].
__global__ void k_edge_agg(const int* __restrict__ eidx,
                           const float* __restrict__ a_s, const float* __restrict__ a_d,
                           const float* __restrict__ m_, const float* __restrict__ s_,
                           const float* __restrict__ x, float* __restrict__ gout) {
  int e = blockIdx.x * blockDim.x + threadIdx.x;
  const int total = E_ + N_;
  if (e >= total) return;
  int src, dst;
  if (e < E_) { src = eidx[e]; dst = eidx[E_ + e]; }
  else { src = e - E_; dst = e - E_; }
#pragma unroll
  for (int h = 0; h < HEADS_; ++h) {
    float v = a_s[(size_t)src * 4 + h] + a_d[(size_t)dst * 4 + h];
    v = (v > 0.f) ? v : 0.2f * v;
    float alpha = __expf(v - m_[(size_t)dst * 4 + h]) / (s_[(size_t)dst * 4 + h] + 1e-16f);
#pragma unroll
    for (int c = 0; c < C_; ++c)
      atomicAdd(&gout[(size_t)dst * 40 + h * C_ + c],
                alpha * x[(size_t)src * 40 + h * C_ + c]);
  }
}

// ---------------------------------------------------------------------------
// Finalize: graph confidence (cosine), combined score, agreement mask.
// ---------------------------------------------------------------------------
__global__ void k_final(const float* __restrict__ probs, const float* __restrict__ gout,
                        const float* __restrict__ gb,
                        const float* __restrict__ deg, const float* __restrict__ nsum,
                        const float* __restrict__ out_lab, const float* __restrict__ out_maxp,
                        const float* __restrict__ out_emb, const float* __restrict__ out_tc,
                        float* __restrict__ out_comb, float* __restrict__ out_mask,
                        float* __restrict__ out_gc) {
  int n = blockIdx.x * blockDim.x + threadIdx.x;
  if (n >= N_) return;
  float g[C_];
#pragma unroll
  for (int c = 0; c < C_; ++c) {
    float a = 0.f;
#pragma unroll
    for (int h = 0; h < HEADS_; ++h) a += gout[(size_t)n * 40 + h * C_ + c];
    g[c] = a * 0.25f + gb[c];
  }
  float num = 0.f, np = 0.f, ng = 0.f;
#pragma unroll
  for (int c = 0; c < C_; ++c) {
    float p = probs[(size_t)n * C_ + c];
    num += p * g[c];
    np += p * p;
    ng += g[c] * g[c];
  }
  float den = fmaxf(sqrtf(np) * sqrtf(ng), 1e-8f);
  float gc = (num / den + 1.0f) * 0.5f;
  out_gc[n] = gc;
  float comb = 0.4f * out_maxp[n] + 0.2f * out_emb[n] + 0.2f * out_tc[n] + 0.2f * gc;
  out_comb[n] = comb;
  int lab = (int)out_lab[n];
  float navg = nsum[(size_t)n * C_ + lab] / (deg[n] + 1e-8f);
  bool fin = (comb > 0.85f) && (navg >= 0.6f);
  out_mask[n] = fin ? 1.0f : 0.0f;
}

// ---------------------------------------------------------------------------
extern "C" void kernel_launch(void* const* d_in, const int* in_sizes, int n_in,
                              void* d_out, int out_size, void* d_ws, size_t ws_size,
                              hipStream_t stream) {
  const float* emb    = (const float*)d_in[0];
  const float* pred   = (const float*)d_in[1];
  // d_in[2] = temporal_features: only its shape matters in the reference
  const int*   eidx   = (const int*)d_in[3];
  const float* cW1    = (const float*)d_in[4];
  const float* cb1    = (const float*)d_in[5];
  const float* cW2    = (const float*)d_in[6];
  const float* cb2    = (const float*)d_in[7];
  const float* w_ih0  = (const float*)d_in[8];
  const float* w_hh0  = (const float*)d_in[9];
  const float* b_ih0  = (const float*)d_in[10];
  const float* b_hh0  = (const float*)d_in[11];
  const float* w_ih1  = (const float*)d_in[12];
  const float* w_hh1  = (const float*)d_in[13];
  const float* b_ih1  = (const float*)d_in[14];
  const float* b_hh1  = (const float*)d_in[15];
  const float* gW     = (const float*)d_in[16];
  const float* ga_src = (const float*)d_in[17];
  const float* ga_dst = (const float*)d_in[18];
  const float* gb     = (const float*)d_in[19];

  float* ws = (float*)d_ws;
  float* probs = ws + OFF_PROBS;
  float* xbuf  = ws + OFF_X;
  float* a_s   = ws + OFF_AS;
  float* a_d   = ws + OFF_AD;
  float* m_    = ws + OFF_M;
  float* s_    = ws + OFF_S;
  float* gout  = ws + OFF_GOUT;
  float* deg   = ws + OFF_DEG;
  float* nsum  = ws + OFF_NSUM;
  _Float16* hbase = (_Float16*)(ws + OFF_F32_END);
  _Float16* cW1h  = hbase + HOFF_CW1;
  _Float16* whh0h = hbase + HOFF_WHH0;
  _Float16* wih1h = hbase + HOFF_WIH1;
  _Float16* whh1h = hbase + HOFF_WHH1;

  float* out = (float*)d_out;
  float* out_lab  = out + 0 * (size_t)N_;
  float* out_comb = out + 1 * (size_t)N_;
  float* out_mask = out + 2 * (size_t)N_;
  float* out_maxp = out + 3 * (size_t)N_;
  float* out_emb  = out + 4 * (size_t)N_;
  float* out_tc   = out + 5 * (size_t)N_;
  float* out_gc   = out + 6 * (size_t)N_;
  float* out_unc  = out + 7 * (size_t)N_;

  // 1. weight conversion to fp16 (L2-resident for the GEMM kernels)
  k_convert_f16<<<(128 * 256 + 255) / 256, 256, 0, stream>>>(cW1, cW1h, 128 * 256);
  k_convert_f16<<<(512 * 128 + 255) / 256, 256, 0, stream>>>(w_hh0, whh0h, 512 * 128);
  k_convert_f16<<<(512 * 128 + 255) / 256, 256, 0, stream>>>(w_ih1, wih1h, 512 * 128);
  k_convert_f16<<<(512 * 128 + 255) / 256, 256, 0, stream>>>(w_hh1, whh1h, 512 * 128);

  // 2. per-node softmax / GAT prep / accumulator init
  k_node<<<(N_ + 255) / 256, 256, 0, stream>>>(pred, gW, ga_src, ga_dst,
                                               probs, xbuf, a_s, a_d, m_, s_, gout,
                                               deg, nsum, out_lab, out_maxp, out_unc);

  // 3. confidence net (WMMA f16 GEMM)
  k_confnet<<<N_ / 16, 256, 0, stream>>>(emb, cb1, cW2, cb2, cW1h, out_emb);

  // 4. 2-layer LSTM (WMMA f16 recurrent GEMMs)
  k_lstm<<<N_ / 16, 256, 0, stream>>>(pred, w_ih0, b_ih0, b_hh0, b_ih1, b_hh1,
                                      whh0h, wih1h, whh1h, out_tc);

  // 5. GAT edge softmax + aggregation (3 passes over E + self loops)
  const int ETOT = E_ + N_;
  k_edge_max<<<(ETOT + 255) / 256, 256, 0, stream>>>(eidx, a_s, a_d, out_lab, m_, deg, nsum);
  k_edge_sum<<<(ETOT + 255) / 256, 256, 0, stream>>>(eidx, a_s, a_d, m_, s_);
  k_edge_agg<<<(ETOT + 255) / 256, 256, 0, stream>>>(eidx, a_s, a_d, m_, s_, xbuf, gout);

  // 6. finalize
  k_final<<<(N_ + 255) / 256, 256, 0, stream>>>(probs, gout, gb, deg, nsum,
                                                out_lab, out_maxp, out_emb, out_tc,
                                                out_comb, out_mask, out_gc);
}
